// GATLayer_11587821765007
// MI455X (gfx1250) — compile-verified, matrix operand order checked by reading, not run on hardware
//
#include <hip/hip_runtime.h>
#include <hip/hip_bf16.h>
#include <math.h>

#define N_NODES   100000
#define N_EDGES   1600000
#define IN_FEATS  256
#define OUT_FEATS 64
#define NHEAD     4
#define TOT_FEATS 256          // OUT_FEATS * NHEAD
#define ALPHA_NS  0.2f
#define M_TILES   (N_NODES / 16)   // 6250
#define K_STEPS   (IN_FEATS / 32)  // 8
#define N_TILES   (TOT_FEATS / 16) // 16

typedef __attribute__((ext_vector_type(16))) __bf16        v16bf;
typedef __attribute__((ext_vector_type(8)))  float         v8f;
typedef __attribute__((ext_vector_type(8)))  unsigned int  v8u;

__device__ __forceinline__ float leaky(float v) { return v > 0.f ? v : ALPHA_NS * v; }

// per-lane K base for 16-bit A/B fragments (ISA 7.12.2): VGPR j, half-wave hi
__device__ __forceinline__ int kbase(int j, int hi) {
  return (j < 4) ? ((hi ? 8 : 0) + 2 * j) : ((hi ? 24 : 16) + 2 * (j - 4));
}

__device__ __forceinline__ unsigned pack_bf16_pair(float lo, float hif) {
  unsigned short u0 = __builtin_bit_cast(unsigned short, (__bf16)lo);
  unsigned short u1 = __builtin_bit_cast(unsigned short, (__bf16)hif);
  return (unsigned)u0 | ((unsigned)u1 << 16);
}

// float atomic-max via sign-split integer atomics (init value must be -inf)
__device__ __forceinline__ void atomicMaxF32(float* addr, float val) {
  if (val >= 0.f) atomicMax((int*)addr, __float_as_int(val));
  else            atomicMin((unsigned int*)addr, __float_as_uint(val));
}

// ---------------- Kernel 1: pack W into per-lane B-fragment layout ----------------
// wpack[((kstep*16 + nTile)*32 + lane)*8 + j] = {W[k,n], W[k+1,n]} as bf16 pair
__global__ void pack_w_kernel(const float* __restrict__ W, unsigned* __restrict__ wpack) {
  int t = blockIdx.x * blockDim.x + threadIdx.x;   // 8*16*32*8 = 32768
  if (t >= K_STEPS * N_TILES * 32 * 8) return;
  int j     = t & 7;
  int lane  = (t >> 3) & 31;
  int nTile = (t >> 8) & 15;
  int kstep = t >> 12;
  int hi = lane >> 4, lid = lane & 15;
  int k = kstep * 32 + kbase(j, hi);
  int n = nTile * 16 + lid;
  wpack[t] = pack_bf16_pair(W[(size_t)k * TOT_FEATS + n], W[(size_t)(k + 1) * TOT_FEATS + n]);
}

// ---------------- Kernel 2: convert+pack x into per-lane A-fragment layout ----------------
// One wave per (mTile, kstep). xpack[((mTile*8 + kstep)*32 + lane)*8 + j]
__global__ void pack_x_kernel(const float* __restrict__ x, unsigned* __restrict__ xpack) {
  int gid  = blockIdx.x * blockDim.x + threadIdx.x;
  int wave = gid >> 5;
  int lane = gid & 31;
  int mTile = wave >> 3;
  int kstep = wave & 7;
  if (mTile >= M_TILES) return;
  int hi = lane >> 4, lid = lane & 15;
  const float* xrow = x + (size_t)(mTile * 16 + lid) * IN_FEATS + kstep * 32;
  v8u pk;
#pragma unroll
  for (int j = 0; j < 8; ++j) {
    float2 f = *(const float2*)(xrow + kbase(j, hi));
    pk[j] = pack_bf16_pair(f.x, f.y);
  }
  *(v8u*)(xpack + ((size_t)(mTile * 8 + kstep) * 32 + lane) * 8) = pk;
}

// ---------------- Kernel 3: h = x @ W via v_wmma_f32_16x16x32_bf16 ----------------
// One wave computes a 16x128 strip: 8 N-tiles share one A fragment per K-step.
__global__ void gemm_kernel(const unsigned* __restrict__ xpack,
                            const unsigned* __restrict__ wpack, float* __restrict__ h) {
  int wave = (blockIdx.x * blockDim.x + threadIdx.x) >> 5;
  int lane = threadIdx.x & 31;
  int mTile = wave >> 1;
  int nHalf = wave & 1;
  if (mTile >= M_TILES) return;
  int hi = lane >> 4, lid = lane & 15;

  v8f acc[8];
#pragma unroll
  for (int q = 0; q < 8; ++q) acc[q] = (v8f){};

  for (int kstep = 0; kstep < K_STEPS; ++kstep) {
    v16bf a = __builtin_bit_cast(
        v16bf, *(const v8u*)(xpack + ((size_t)(mTile * 8 + kstep) * 32 + lane) * 8));
#pragma unroll
    for (int q = 0; q < 8; ++q) {
      int nTile = nHalf * 8 + q;
      v16bf b = __builtin_bit_cast(
          v16bf, *(const v8u*)(wpack + ((size_t)(kstep * 16 + nTile) * 32 + lane) * 8));
      acc[q] = __builtin_amdgcn_wmma_f32_16x16x32_bf16(false, a, false, b, (short)0,
                                                       acc[q], false, false);
    }
  }

  int rowBase = mTile * 16 + hi * 8;   // D tile: row = v + 8*hi, col = lid
#pragma unroll
  for (int q = 0; q < 8; ++q) {
    float* op = h + (size_t)rowBase * TOT_FEATS + (nHalf * 8 + q) * 16 + lid;
#pragma unroll
    for (int v = 0; v < 8; ++v) {
      float val = acc[q][v];
      val = (val != val) ? 0.f : val;   // jnp.where(isnan(h), 0, h)
      op[(size_t)v * TOT_FEATS] = val;
    }
  }
}

// ---------------- Kernel 4: per-node attention logits ----------------
__global__ void logits_kernel(const float* __restrict__ h, const float* __restrict__ a_l,
                              const float* __restrict__ a_r, float* __restrict__ hl,
                              float* __restrict__ hr) {
  int t = blockIdx.x * blockDim.x + threadIdx.x;
  if (t >= N_NODES * NHEAD) return;
  int node = t >> 2, hd = t & 3;
  const float* hp = h + (size_t)node * TOT_FEATS + hd * OUT_FEATS;
  const float* al = a_l + hd * OUT_FEATS;
  const float* ar = a_r + hd * OUT_FEATS;
  float sl = 0.f, sr = 0.f;
#pragma unroll 4
  for (int f = 0; f < OUT_FEATS; ++f) {
    float v = hp[f];
    sl += al[f] * v;
    sr += ar[f] * v;
  }
  hl[t] = sl;
  hr[t] = sr;
}

// ---------------- Kernel 5: init out/max/denom ----------------
__global__ void init_kernel(float* __restrict__ out, float* __restrict__ mx,
                            float* __restrict__ dn) {
  size_t t = (size_t)blockIdx.x * blockDim.x + threadIdx.x;
  if (t < (size_t)N_NODES * TOT_FEATS) out[t] = 0.f;
  if (t < (size_t)N_NODES * NHEAD) { mx[t] = -INFINITY; dn[t] = 0.f; }
}

// ---------------- Kernel 6: segment max over edges ----------------
__global__ void edge_max_kernel(const int* __restrict__ row, const int* __restrict__ col,
                                const float* __restrict__ hl, const float* __restrict__ hr,
                                float* __restrict__ mx) {
  int t = blockIdx.x * blockDim.x + threadIdx.x;
  if (t >= N_EDGES * NHEAD) return;
  int e = t >> 2, hd = t & 3;
  int r = row[e], c = col[e];
  float v = leaky(hl[r * NHEAD + hd] + hr[c * NHEAD + hd]);
  atomicMaxF32(&mx[r * NHEAD + hd], v);
}

// ---------------- Kernel 6b: empty segments (-inf) -> 0 ----------------
__global__ void fix_max_kernel(float* __restrict__ mx) {
  int t = blockIdx.x * blockDim.x + threadIdx.x;
  if (t >= N_NODES * NHEAD) return;
  float v = mx[t];
  if (!(v - v == 0.f)) mx[t] = 0.f;   // non-finite -> 0
}

// ---------------- Kernel 7: segment sum of exp ----------------
__global__ void edge_sum_kernel(const int* __restrict__ row, const int* __restrict__ col,
                                const float* __restrict__ hl, const float* __restrict__ hr,
                                const float* __restrict__ mx, float* __restrict__ dn) {
  int t = blockIdx.x * blockDim.x + threadIdx.x;
  if (t >= N_EDGES * NHEAD) return;
  int e = t >> 2, hd = t & 3;
  int r = row[e], c = col[e];
  float v  = leaky(hl[r * NHEAD + hd] + hr[c * NHEAD + hd]);
  float ex = __expf(v - mx[r * NHEAD + hd]);
  unsafeAtomicAdd(&dn[r * NHEAD + hd], ex);
}

// ---------------- Kernel 8: sparse weighted aggregation ----------------
// One wave per edge; lane owns 8 contiguous features (head = lane/8).
__global__ void aggregate_kernel(const int* __restrict__ row, const int* __restrict__ col,
                                 const float* __restrict__ h, const float* __restrict__ hl,
                                 const float* __restrict__ hr, const float* __restrict__ mx,
                                 const float* __restrict__ dn, float* __restrict__ out) {
  size_t gid = (size_t)blockIdx.x * blockDim.x + threadIdx.x;
  size_t e = gid >> 5;
  if (e >= N_EDGES) return;
  int lane = (int)(gid & 31);
  int r = row[e], c = col[e];
  int hd = lane >> 3;
  float ev  = leaky(hl[r * NHEAD + hd] + hr[c * NHEAD + hd]);
  float ex  = __expf(ev - mx[r * NHEAD + hd]);
  float att = ex / (dn[r * NHEAD + hd] + 1e-16f);

  const float4* hp = (const float4*)(h + (size_t)c * TOT_FEATS + lane * 8);
  float* op = out + (size_t)r * TOT_FEATS + lane * 8;
  float4 v0 = hp[0], v1 = hp[1];
  unsafeAtomicAdd(op + 0, att * v0.x);
  unsafeAtomicAdd(op + 1, att * v0.y);
  unsafeAtomicAdd(op + 2, att * v0.z);
  unsafeAtomicAdd(op + 3, att * v0.w);
  unsafeAtomicAdd(op + 4, att * v1.x);
  unsafeAtomicAdd(op + 5, att * v1.y);
  unsafeAtomicAdd(op + 6, att * v1.z);
  unsafeAtomicAdd(op + 7, att * v1.w);
}

extern "C" void kernel_launch(void* const* d_in, const int* in_sizes, int n_in,
                              void* d_out, int out_size, void* d_ws, size_t ws_size,
                              hipStream_t stream) {
  const float* x   = (const float*)d_in[0];
  const float* W   = (const float*)d_in[1];
  const float* a_l = (const float*)d_in[2];
  const float* a_r = (const float*)d_in[3];
  const int*   row = (const int*)d_in[4];
  const int*   col = (const int*)d_in[5];
  float* out = (float*)d_out;

  // workspace carve-out (~160 MB)
  char* ws = (char*)d_ws;
  size_t off = 0;
  unsigned* wpack = (unsigned*)(ws + off);
  off += (size_t)K_STEPS * N_TILES * 32 * 8 * sizeof(unsigned);          // 128 KB
  unsigned* xpack = (unsigned*)(ws + off);
  off += (size_t)M_TILES * K_STEPS * 32 * 8 * sizeof(unsigned);          // 51.2 MB
  float* h  = (float*)(ws + off); off += (size_t)N_NODES * TOT_FEATS * sizeof(float);
  float* hl = (float*)(ws + off); off += (size_t)N_NODES * NHEAD * sizeof(float);
  float* hr = (float*)(ws + off); off += (size_t)N_NODES * NHEAD * sizeof(float);
  float* mx = (float*)(ws + off); off += (size_t)N_NODES * NHEAD * sizeof(float);
  float* dn = (float*)(ws + off); off += (size_t)N_NODES * NHEAD * sizeof(float);

  pack_w_kernel<<<(K_STEPS * N_TILES * 32 * 8 + 255) / 256, 256, 0, stream>>>(W, wpack);

  int packx_threads = M_TILES * K_STEPS * 32;                            // 1.6M
  pack_x_kernel<<<(packx_threads + 255) / 256, 256, 0, stream>>>(x, xpack);

  int gemm_threads = M_TILES * 2 * 32;                                   // 400000
  gemm_kernel<<<(gemm_threads + 255) / 256, 256, 0, stream>>>(xpack, wpack, h);

  logits_kernel<<<(N_NODES * NHEAD + 255) / 256, 256, 0, stream>>>(h, a_l, a_r, hl, hr);

  size_t init_elems = (size_t)N_NODES * TOT_FEATS;
  init_kernel<<<(unsigned)((init_elems + 255) / 256), 256, 0, stream>>>(out, mx, dn);

  edge_max_kernel<<<(N_EDGES * NHEAD + 255) / 256, 256, 0, stream>>>(row, col, hl, hr, mx);
  fix_max_kernel<<<(N_NODES * NHEAD + 255) / 256, 256, 0, stream>>>(mx);
  edge_sum_kernel<<<(N_EDGES * NHEAD + 255) / 256, 256, 0, stream>>>(row, col, hl, hr, mx, dn);

  size_t agg_threads = (size_t)N_EDGES * 32;
  aggregate_kernel<<<(unsigned)((agg_threads + 255) / 256), 256, 0, stream>>>(
      row, col, h, hl, hr, mx, dn, out);
}
